// GatedMultiScaleRetentionLayer_32624571581026
// MI455X (gfx1250) — matrix operand order repaired and checked by reading, not compiled, not run on hardware
//
#include <hip/hip_runtime.h>
#include <hip/hip_bf16.h>
#include <math.h>

// ---------------- problem constants ----------------
#define B_DIM   2
#define S_LEN   2048
#define D_MODEL 1024
#define NHEADS  16
#define HDIM    64
#define BS      (B_DIM * S_LEN)          // 4096 rows
#define BH      (B_DIM * NHEADS)         // 32 (b,h) pairs
#define BHSD    (BH * S_LEN * HDIM)      // 4,194,304 elements

typedef __attribute__((ext_vector_type(16))) _Float16     v16h;
typedef __attribute__((ext_vector_type(8)))  _Float16     v8h;
typedef __attribute__((ext_vector_type(8)))  float        v8f;
typedef __attribute__((ext_vector_type(4)))  unsigned int v4u;
typedef __attribute__((ext_vector_type(8)))  int          v8i;
typedef __attribute__((ext_vector_type(4)))  int          v4i;

static __device__ inline v8f zero_v8f() { v8f z = {}; return z; }

// Build a v16h A/B fragment from two 16-byte chunks (keeps loads b128-aligned).
static __device__ inline v16h ld16h(const _Float16* p0, const _Float16* p1) {
  v8h lo = *(const v8h*)p0;
  v8h hi = *(const v8h*)p1;
  v16h r;
#pragma unroll
  for (int i = 0; i < 8; ++i) { r[i] = lo[i]; r[i + 8] = hi[i]; }
  return r;
}

static __device__ inline v8f wmma16(v16h a, v16h b, v8f c) {
  // (neg_a, A, neg_b, B, c_mod, C, reuse_a, reuse_b)
  return __builtin_amdgcn_wmma_f32_16x16x32_f16(false, a, false, b, (short)0, c,
                                                false, false);
}

// ---- Tensor Data Mover: DMA a 2D f16 tile (32 cols x 128 rows, row stride
// 1024 elements) from global memory into LDS. Descriptor per cdna5 D# spec.
static __device__ inline void tdm_load_tile(const _Float16* gsrc,
                                            unsigned lds_off) {
  unsigned long long ga = (unsigned long long)(uintptr_t)gsrc;
  v4u g0;
  g0[0] = 1u;                                   // count=1 (valid user desc)
  g0[1] = lds_off;                              // lds_addr (bytes)
  g0[2] = (unsigned)(ga & 0xFFFFFFFFu);         // global_addr[31:0]
  g0[3] = (unsigned)((ga >> 32) & 0x01FFFFFFu)  // global_addr[56:32]
          | (2u << 30);                         // type=2 ("image")
  v8i g1;
  g1[0] = (int)(1u << 16);                      // wg_mask=0, data_size=1 (2B)
  g1[1] = (int)((1024u & 0xFFFFu) << 16);       // tensor_dim0[15:0] @ [63:48]
  g1[2] = (int)((1024u >> 16) |                 // tensor_dim0[31:16]
                ((1024u & 0xFFFFu) << 16));     // tensor_dim1[15:0]
  g1[3] = (int)((1024u >> 16) | (32u << 16));   // tensor_dim1[31:16]|tile_dim0
  g1[4] = 128;                                  // tile_dim1=128, tile_dim2=0
  g1[5] = 1024;                                 // tensor_dim0_stride[31:0]
  g1[6] = 0;                                    // stride hi | dim1_stride lo
  g1[7] = 0;
  v4i gz = {0, 0, 0, 0};
#if defined(__clang_major__) && (__clang_major__ >= 23)
  v8i gz8 = {0, 0, 0, 0, 0, 0, 0, 0};
  __builtin_amdgcn_tensor_load_to_lds(g0, g1, gz, gz, gz8, 0);
#else
  __builtin_amdgcn_tensor_load_to_lds(g0, g1, gz, gz, 0);
#endif
}

// ---------------- small utility kernels ----------------
__global__ void zero_stats_kernel(float* stats) {
  int i = threadIdx.x;
  if (i < 2 * BH) stats[i] = 0.0f;
}

__global__ void cvt_f32_to_f16_kernel(const float* __restrict__ in,
                                      _Float16* __restrict__ out, int n) {
  int i = blockIdx.x * blockDim.x + threadIdx.x;
  if (i < n) out[i] = (_Float16)in[i];
}

// W (K x N, f32, row-major) -> Wt (N x K, f16, row-major)
__global__ void transpose_w_kernel(const float* __restrict__ W,
                                   _Float16* __restrict__ Wt) {
  int i = blockIdx.x * blockDim.x + threadIdx.x;  // over D*D
  if (i >= D_MODEL * D_MODEL) return;
  int n = i / D_MODEL, k = i % D_MODEL;
  Wt[(size_t)n * D_MODEL + k] = (_Float16)W[(size_t)k * D_MODEL + n];
}

// ---------------- WMMA GEMM with TDM-staged B tiles ----------------
// C(MxN,f32) = A(MxK,f16) @ Wt^T + bias, Wt is (N x K) f16 row-major.
// Block = 256 threads = 8 waves (4 M-waves x 2 N-waves), block tile 64x128.
// B tile (128 rows x 32 K) is DMA'd into LDS by the Tensor Data Mover,
// double-buffered so DMA of chunk i+1 overlaps WMMA of chunk i.
__global__ __launch_bounds__(256)
void gemm_f16_kernel(const _Float16* __restrict__ A,
                     const _Float16* __restrict__ Wt,
                     const float* __restrict__ bias,
                     float* __restrict__ C, int M, int N, int K) {
  __shared__ _Float16 sB[2][128 * 32];

  const int tid  = threadIdx.x;
  const int lane = tid & 31;
  const int wid  = tid >> 5;
  const int ln   = lane & 15;   // N-col / M-row selector within fragment
  const int hf   = lane >> 4;   // K half selector

  const int mbase  = blockIdx.x * 64 + (wid & 3) * 16;
  const int nblk   = blockIdx.y * 128;           // block N base (TDM tile)
  const int wnoff  = (wid >> 2) * 64;            // this wave's N offset in tile

  const _Float16* arow  = A + (size_t)(mbase + ln) * K;
  const _Float16* btile = Wt + (size_t)nblk * K; // row stride K elements

  const unsigned ldsOff[2] = {(unsigned)(uintptr_t)&sB[0][0],
                              (unsigned)(uintptr_t)&sB[1][0]};

  v8f acc[4];
#pragma unroll
  for (int nb = 0; nb < 4; ++nb) acc[nb] = zero_v8f();

  const int nchunk = K >> 5;
  if (wid == 0) tdm_load_tile(btile, ldsOff[0]);

  for (int i = 0; i < nchunk; ++i) {
    const int k0 = i << 5;
    if (wid == 0) {
      if (i + 1 < nchunk) {
        tdm_load_tile(btile + (k0 + 32), ldsOff[(i + 1) & 1]);
        __builtin_amdgcn_s_wait_tensorcnt(1);   // chunk i landed
      } else {
        __builtin_amdgcn_s_wait_tensorcnt(0);
      }
    }
    __syncthreads();                            // publish LDS tile

    if (k0 + 128 < K)
      __builtin_prefetch((const void*)(arow + k0 + 128), 0, 3);
    // A fragment: lane (ln,hf) holds K run [k0+16*hf,+16)
    v16h af = ld16h(arow + k0 + 16 * hf, arow + k0 + 16 * hf + 8);

    const _Float16* bb = &sB[i & 1][0];
#pragma unroll
    for (int nb = 0; nb < 4; ++nb) {
      const _Float16* brow = bb + (size_t)(wnoff + nb * 16 + ln) * 32 + 16 * hf;
      v16h bf = ld16h(brow, brow + 8);
      acc[nb] = wmma16(af, bf, acc[nb]);
    }
    __syncthreads();                            // tile consumed, reusable
  }

  // C/D layout: VGPR r, lane -> (M = r + 8*hf, N = ln)
#pragma unroll
  for (int nb = 0; nb < 4; ++nb) {
    int col = nblk + wnoff + nb * 16 + ln;
    float bia = bias[col];
#pragma unroll
    for (int r = 0; r < 8; ++r) {
      int row = mbase + r + 8 * hf;
      C[(size_t)row * N + col] = acc[nb][r] + bia;
    }
  }
}

// ---------------- rotary xPos: q,k f32 (B,S,D) -> f16 (B,H,S,HDIM) ----------
__global__ void rotary_kernel(const float* __restrict__ qf,
                              const float* __restrict__ kf,
                              _Float16* __restrict__ qh,
                              _Float16* __restrict__ kh) {
  int idx = blockIdx.x * blockDim.x + threadIdx.x;  // over BHSD
  if (idx >= BHSD) return;
  int dh = idx & (HDIM - 1);
  int r  = idx >> 6;
  int s  = r % S_LEN;
  int r2 = r / S_LEN;
  int h  = r2 % NHEADS;
  int b  = r2 / NHEADS;

  size_t inIdx = ((size_t)(b * S_LEN + s)) * D_MODEL + h * HDIM + dh;
  float qv = qf[inIdx], kv = kf[inIdx];
  float qo, ko;
  if (dh < 32) {
    int p = dh >> 1;
    float t    = (float)s;
    float invf = powf(10000.0f, -(float)(2 * p) / 32.0f);
    float ang  = t * invf;
    float cv = cosf(ang), sv = sinf(ang);
    float base  = ((float)(2 * p) + 0.4f * 32.0f) / (1.4f * 32.0f);
    float power = (t - (float)(S_LEN / 2)) / 512.0f;
    float sc  = powf(base, power);
    float isc = 1.0f / sc;
    size_t pIdx = inIdx - dh + (dh ^ 1);
    float qp = qf[pIdx], kp = kf[pIdx];
    float qrot = (dh & 1) ? qp : -qp;  // rotate_half
    float krot = (dh & 1) ? kp : -kp;
    qo = qv * (cv * sc) + qrot * (sv * sc);
    ko = kv * (cv * isc) + krot * (sv * isc);
  } else {
    qo = qv; ko = kv;
  }
  size_t outIdx = (((size_t)(b * NHEADS + h)) * S_LEN + s) * HDIM + dh;
  qh[outIdx] = (_Float16)qo;
  kh[outIdx] = (_Float16)ko;
}

// v f32 (B,S,D) -> vT f16 (B,H,HDIM,S)
__global__ void transpose_v_kernel(const float* __restrict__ vf,
                                   _Float16* __restrict__ vT) {
  int idx = blockIdx.x * blockDim.x + threadIdx.x;  // over BHSD
  if (idx >= BHSD) return;
  int dh = idx & (HDIM - 1);
  int r  = idx >> 6;
  int s  = r % S_LEN;
  int r2 = r / S_LEN;
  int h  = r2 % NHEADS;
  int b  = r2 / NHEADS;
  float v = vf[((size_t)(b * S_LEN + s)) * D_MODEL + h * HDIM + dh];
  vT[(((size_t)(b * NHEADS + h)) * HDIM + dh) * S_LEN + s] = (_Float16)v;
}

// ---------------- retention core (flash-style, decay in closed form) -------
// 1 wave / block; block = 16 query rows of one (b,h).
__global__ __launch_bounds__(32)
void retention_kernel(const _Float16* __restrict__ qh,
                      const _Float16* __restrict__ kh,
                      const _Float16* __restrict__ vT,
                      float* __restrict__ ret,
                      float* __restrict__ stats) {
  __shared__ _Float16 sA[16 * 32];  // score tile relayout C->A

  const int bh    = blockIdx.y;       // 0..31
  const int h     = bh % NHEADS;
  const int qbase = blockIdx.x * 16;
  const int lane  = threadIdx.x;
  const int ln    = lane & 15;
  const int hf    = lane >> 4;

  const float gamma = 1.0f - exp2f(-5.0f - (float)h);
  const float lg    = log2f(gamma);

  const _Float16* qB = qh + (size_t)bh * S_LEN * HDIM;
  const _Float16* kB = kh + (size_t)bh * S_LEN * HDIM;
  const _Float16* vB = vT + (size_t)bh * HDIM * S_LEN;

  // q A-fragments for K = 0..31 and 32..63
  v16h qa[2];
  {
    const _Float16* row = qB + (size_t)(qbase + ln) * HDIM;
#pragma unroll
    for (int sl = 0; sl < 2; ++sl)
      qa[sl] = ld16h(row + 32 * sl + 16 * hf, row + 32 * sl + 16 * hf + 8);
  }

  v8f acc[4];
#pragma unroll
  for (int nb = 0; nb < 4; ++nb) acc[nb] = zero_v8f();

  for (int j0 = 0; j0 <= qbase; j0 += 32) {
    // ---- scores: two 16x16 tiles, K = HDIM = 64 (2 chained WMMAs each) ----
    v8f sc0 = zero_v8f(), sc1 = zero_v8f();
#pragma unroll
    for (int sl = 0; sl < 2; ++sl) {
      const _Float16* kr0 = kB + (size_t)(j0 + ln) * HDIM + 32 * sl + 16 * hf;
      const _Float16* kr1 = kB + (size_t)(j0 + 16 + ln) * HDIM + 32 * sl + 16 * hf;
      sc0 = wmma16(qa[sl], ld16h(kr0, kr0 + 8), sc0);
      sc1 = wmma16(qa[sl], ld16h(kr1, kr1 + 8), sc1);
    }
    // ---- decay mask gamma^(i-j) (0 above diagonal) + C->A relayout in LDS --
#pragma unroll
    for (int t = 0; t < 2; ++t) {
      int j = j0 + 16 * t + ln;
#pragma unroll
      for (int r = 0; r < 8; ++r) {
        int i  = qbase + r + 8 * hf;
        int dd = i - j;
        float f = (dd < 0) ? 0.0f : exp2f((float)dd * lg);
        float sv = ((t == 0) ? sc0[r] : sc1[r]) * f;
        sA[(r + 8 * hf) * 32 + 16 * t + ln] = (_Float16)sv;
      }
    }
    __syncthreads();
    // A fragment of the 16x32 score tile
    v16h af = ld16h(&sA[ln * 32 + 16 * hf], &sA[ln * 32 + 16 * hf + 8]);
    __syncthreads();
    // ---- ret(16x64) += scores(16x32) @ v(32x64), vT rows contiguous in S --
#pragma unroll
    for (int nb = 0; nb < 4; ++nb) {
      const _Float16* vr =
          vB + (size_t)(nb * 16 + ln) * S_LEN + j0 + 16 * hf;
      acc[nb] = wmma16(af, ld16h(vr, vr + 8), acc[nb]);
    }
  }

  // ---- row normalization 1/sqrt(sum_p gamma^p), store, groupnorm stats ----
  float psum = 0.0f, psq = 0.0f;
#pragma unroll
  for (int r = 0; r < 8; ++r) {
    int i = qbase + r + 8 * hf;
    float sg = (1.0f - exp2f((float)(i + 1) * lg)) * exp2f(5.0f + (float)h);
    float rn = rsqrtf(sg);
#pragma unroll
    for (int nb = 0; nb < 4; ++nb) {
      float v = acc[nb][r] * rn;
      ret[((size_t)bh * S_LEN + i) * HDIM + nb * 16 + ln] = v;
      psum += v;
      psq  += v * v;
    }
  }
#pragma unroll
  for (int off = 16; off > 0; off >>= 1) {
    psum += __shfl_xor(psum, off, 32);
    psq  += __shfl_xor(psq, off, 32);
  }
  if (lane == 0) {
    atomicAdd(&stats[bh * 2 + 0], psum);
    atomicAdd(&stats[bh * 2 + 1], psq);
  }
}

// ---------------- groupnorm + silu gate fusion -> f16 ----------------------
__global__ void gatenorm_kernel(const float* __restrict__ retv,
                                const float* __restrict__ gf,
                                const float* __restrict__ stats,
                                const float* __restrict__ gn_w,
                                const float* __restrict__ gn_b,
                                _Float16* __restrict__ hbuf) {
  int idx = blockIdx.x * blockDim.x + threadIdx.x;  // over BS*D
  if (idx >= BS * D_MODEL) return;
  int col = idx % D_MODEL;
  int h   = col >> 6;
  int dh  = col & (HDIM - 1);
  int row = idx / D_MODEL;
  int s   = row % S_LEN;
  int b   = row / S_LEN;
  int bh  = b * NHEADS + h;

  const float cnt  = (float)(S_LEN * HDIM);
  float mean = stats[bh * 2 + 0] / cnt;
  float var  = stats[bh * 2 + 1] / cnt - mean * mean;
  float x    = retv[(((size_t)bh) * S_LEN + s) * HDIM + dh];
  float nrm  = (x - mean) * rsqrtf(var + 1e-5f) * gn_w[h] + gn_b[h];
  float g    = gf[idx];
  float silu = g / (1.0f + expf(-g));
  hbuf[idx]  = (_Float16)(silu * nrm);
}

// ---------------- launcher ----------------
extern "C" void kernel_launch(void* const* d_in, const int* in_sizes, int n_in,
                              void* d_out, int out_size, void* d_ws,
                              size_t ws_size, hipStream_t stream) {
  const float* x    = (const float*)d_in[0];
  const float* Wq   = (const float*)d_in[1];
  const float* bq   = (const float*)d_in[2];
  const float* Wk   = (const float*)d_in[3];
  const float* bk   = (const float*)d_in[4];
  const float* Wv   = (const float*)d_in[5];
  const float* bv   = (const float*)d_in[6];
  const float* Wg   = (const float*)d_in[7];
  const float* bg   = (const float*)d_in[8];
  const float* Wo   = (const float*)d_in[9];
  const float* bo   = (const float*)d_in[10];
  const float* gn_w = (const float*)d_in[11];
  const float* gn_b = (const float*)d_in[12];
  float* out = (float*)d_out;

  // carve workspace (256B aligned)
  size_t off = 0;
  char* base = (char*)d_ws;
  auto carve = [&](size_t bytes) -> void* {
    void* p = base + off;
    off += (bytes + 255) & ~(size_t)255;
    return p;
  };
  _Float16* xh   = (_Float16*)carve((size_t)BS * D_MODEL * 2);
  _Float16* WqT  = (_Float16*)carve((size_t)D_MODEL * D_MODEL * 2);
  _Float16* WkT  = (_Float16*)carve((size_t)D_MODEL * D_MODEL * 2);
  _Float16* WvT  = (_Float16*)carve((size_t)D_MODEL * D_MODEL * 2);
  _Float16* WgT  = (_Float16*)carve((size_t)D_MODEL * D_MODEL * 2);
  _Float16* WoT  = (_Float16*)carve((size_t)D_MODEL * D_MODEL * 2);
  float*    qf   = (float*)carve((size_t)BS * D_MODEL * 4);
  float*    kf   = (float*)carve((size_t)BS * D_MODEL * 4);
  float*    vf   = (float*)carve((size_t)BS * D_MODEL * 4);
  float*    gf   = (float*)carve((size_t)BS * D_MODEL * 4);
  _Float16* qhh  = (_Float16*)carve((size_t)BHSD * 2);
  _Float16* khh  = (_Float16*)carve((size_t)BHSD * 2);
  _Float16* vTh  = (_Float16*)carve((size_t)BHSD * 2);
  float*    retb = (float*)carve((size_t)BHSD * 4);
  _Float16* hbuf = (_Float16*)carve((size_t)BS * D_MODEL * 2);
  float*    stat = (float*)carve(2 * BH * 4);
  (void)ws_size; (void)in_sizes; (void)n_in; (void)out_size;

  const int thr = 256;
  const int nElemBSD = BS * D_MODEL;     // 4,194,304
  const int nBlkBSD  = nElemBSD / thr;   // 16384
  const int nBlkDD   = (D_MODEL * D_MODEL) / thr;  // 4096
  dim3 gemmGrid(BS / 64, D_MODEL / 128);

  zero_stats_kernel<<<1, 64, 0, stream>>>(stat);
  cvt_f32_to_f16_kernel<<<nBlkBSD, thr, 0, stream>>>(x, xh, nElemBSD);
  transpose_w_kernel<<<nBlkDD, thr, 0, stream>>>(Wq, WqT);
  transpose_w_kernel<<<nBlkDD, thr, 0, stream>>>(Wk, WkT);
  transpose_w_kernel<<<nBlkDD, thr, 0, stream>>>(Wv, WvT);
  transpose_w_kernel<<<nBlkDD, thr, 0, stream>>>(Wg, WgT);
  transpose_w_kernel<<<nBlkDD, thr, 0, stream>>>(Wo, WoT);

  gemm_f16_kernel<<<gemmGrid, thr, 0, stream>>>(xh, WqT, bq, qf, BS, D_MODEL, D_MODEL);
  gemm_f16_kernel<<<gemmGrid, thr, 0, stream>>>(xh, WkT, bk, kf, BS, D_MODEL, D_MODEL);
  gemm_f16_kernel<<<gemmGrid, thr, 0, stream>>>(xh, WvT, bv, vf, BS, D_MODEL, D_MODEL);
  gemm_f16_kernel<<<gemmGrid, thr, 0, stream>>>(xh, WgT, bg, gf, BS, D_MODEL, D_MODEL);

  rotary_kernel<<<nBlkBSD, thr, 0, stream>>>(qf, kf, qhh, khh);
  transpose_v_kernel<<<nBlkBSD, thr, 0, stream>>>(vf, vTh);

  dim3 retGrid(S_LEN / 16, BH);
  retention_kernel<<<retGrid, 32, 0, stream>>>(qhh, khh, vTh, retb, stat);

  gatenorm_kernel<<<nBlkBSD, thr, 0, stream>>>(retb, gf, stat, gn_w, gn_b, hbuf);

  gemm_f16_kernel<<<gemmGrid, thr, 0, stream>>>(hbuf, WoT, bo, out, BS, D_MODEL, D_MODEL);
}
